// EnhancedSTGCN_38147899523377
// MI455X (gfx1250) — matrix operand order repaired and checked by reading, not compiled, not run on hardware
//
#include <hip/hip_runtime.h>
#include <hip/hip_bf16.h>

typedef __attribute__((ext_vector_type(16))) _Float16 v16h;
typedef __attribute__((ext_vector_type(8)))  float    v8f;

#define EPSN 1e-5f
#define SM_A 0.85f

#define LDS_KSTRIDE 40   // halves per column slot (32 + pad; 80B keeps b128 16B-aligned, conflict-free loads)

__device__ __forceinline__ float bnscale(float g) { return g * rsqrtf(1.0f + EPSN); }

// ---------------------------------------------------------------------------
// data_bn: xbn[n,c,t,v] = x * (g[c*V+v]/sqrt(1+eps)) + b[c*V+v]
// ---------------------------------------------------------------------------
__global__ void k_databn(const float* __restrict__ x, const float* __restrict__ g,
                         const float* __restrict__ b, float* __restrict__ xbn,
                         int N, int C, int T, int V) {
    int idx = blockIdx.x * blockDim.x + threadIdx.x;
    int total = N * C * T * V;
    if (idx >= total) return;
    int v = idx % V;
    int c = (idx / (T * V)) % C;
    int f = c * V + v;
    xbn[idx] = x[idx] * bnscale(g[f]) + b[f];
}

__global__ void k_w2h(const float* __restrict__ in, _Float16* __restrict__ out, int n) {
    int i = blockIdx.x * blockDim.x + threadIdx.x;
    if (i < n) out[i] = (_Float16)in[i];
}

// ---------------------------------------------------------------------------
// block1 front: graph conv over V (C=2), channel mix 2->64, bn1, relu -> f16 u1
// ---------------------------------------------------------------------------
__global__ void k_gc1(const float* __restrict__ xbn, const float* __restrict__ A,
                      const float* __restrict__ gw, const float* __restrict__ gb,
                      const float* __restrict__ bn1g, const float* __restrict__ bn1b,
                      _Float16* __restrict__ u1, int N, int T, int V, int Co) {
    int idx = blockIdx.x * blockDim.x + threadIdx.x;
    int total = N * T * V;
    if (idx >= total) return;
    int w = idx % V;
    int t = (idx / V) % T;
    int n = idx / (V * T);
    float xa0 = 0.0f, xa1 = 0.0f;
    const float* x0 = xbn + (((size_t)n * 2 + 0) * T + t) * V;
    const float* x1 = xbn + (((size_t)n * 2 + 1) * T + t) * V;
    for (int v = 0; v < V; ++v) {
        float aw = A[w * V + v];
        xa0 += x0[v] * aw;
        xa1 += x1[v] * aw;
    }
    for (int o = 0; o < Co; ++o) {
        float y = gw[o * 2 + 0] * xa0 + gw[o * 2 + 1] * xa1 + gb[o];
        y = y * bnscale(bn1g[o]) + bn1b[o];
        u1[(((size_t)n * Co + o) * T + t) * V + w] = (_Float16)fmaxf(y, 0.0f);
    }
}

// ---------------------------------------------------------------------------
// graph conv over joints: gtmp[n,c,t,w] = sum_v h[n,c,t,v]*A[w,v]  (f32 -> f16)
// ---------------------------------------------------------------------------
__global__ void k_gjoint(const float* __restrict__ h, const float* __restrict__ A,
                         _Float16* __restrict__ gtmp, int N, int C, int T, int V) {
    int idx = blockIdx.x * blockDim.x + threadIdx.x;
    int total = N * C * T * V;
    if (idx >= total) return;
    int w = idx % V;
    size_t rowbase = (size_t)(idx / V) * V;
    float acc = 0.0f;
    const float* hr = h + rowbase;
    const float* Ar = A + w * V;
    for (int v = 0; v < V; ++v) acc += hr[v] * Ar[v];
    gtmp[idx] = (_Float16)acc;
}

// ---------------------------------------------------------------------------
// LDS-staged WMMA implicit GEMM: out[m,col] = epi( sum_k W[m,k]*B[k,col] )
//   col = (n*Tout + t)*V + v ; block tile 64M x 128N, K-chunk 32, double buffer
// All gather indices are 32-bit offsets from uniform SGPR bases (SADDR form),
// zero-padding is branchless (clamped index + select).
// BMODE: 0 = 9-tap temporal conv (f16 act, pad 4, stride)
//        1 = 1x1 over f16 act
//        2 = 1x1 over f32 act, temporal stride
// EPI:   0 = bn2 + inline 2-ch residual from xbn + relu -> f32
//        1 = bn2 + residual f16 buffer + relu -> f32
//        2 = bn1 + relu -> f16
//        3 = + bias -> f16
// ---------------------------------------------------------------------------
template <int BMODE, int EPI>
__global__ __launch_bounds__(256) void wmma_gemm(
    const _Float16* __restrict__ Wh, const _Float16* __restrict__ Bh,
    const float* __restrict__ Bf, float* __restrict__ outF,
    _Float16* __restrict__ outH, const float* __restrict__ bias,
    const float* __restrict__ bng, const float* __restrict__ bnb,
    const _Float16* __restrict__ resH, const float* __restrict__ rw2,
    const float* __restrict__ rb2, const float* __restrict__ xbn,
    int M, int K, int Ci, int Tin, int Tout, int V, int stride) {
    __shared__ _Float16 Blds[2][128 * LDS_KSTRIDE];   // 2 x 10KB

    const int tid    = threadIdx.x;
    const int lane   = tid & 31;
    const int wid    = tid >> 5;          // 8 waves; wave w owns cols [w*16, w*16+16)
    const int group  = lane >> 4;
    const int ncol16 = lane & 15;

    const int cs = Tin * V;               // channel stride (elements), fits 32-bit

    // ---- fill-side: fixed column slot per thread, decoded once ----
    const int fcol  = tid & 127;          // column within block (constant across passes)
    const int krow0 = tid >> 7;           // 0/1 ; k_local = krow0 + 2*pass
    const int fcolg = blockIdx.x * 128 + fcol;
    const int fv    = fcolg % V;
    const int frest = fcolg / V;
    const int ft    = frest % Tout;
    const int fn    = frest / Tout;
    const int ftb   = ft * stride - 4;    // 9-tap window base
    // 32-bit base element offsets (k-independent part)
    const int fbase0 = fn * Ci * cs + ftb * V + fv;            // BMODE 0
    const int fbase1 = fn * Ci * cs + ft * V + fv;             // BMODE 1
    const int fbase2 = fn * Ci * cs + (ft * stride) * V + fv;  // BMODE 2
    _Float16* lslot0 = &Blds[0][fcol * LDS_KSTRIDE + krow0];
    _Float16* lslot1 = &Blds[1][fcol * LDS_KSTRIDE + krow0];

    // ---- consume-side: this lane's output column, decoded once ----
    const int col  = blockIdx.x * 128 + wid * 16 + ncol16;
    const int v    = col % V;
    const int rest = col / V;
    const int t    = rest % Tout;
    const int n    = rest / Tout;

    // A fragments: 4 M-tiles per wave, row = mblock + mt*16 + (lane&15);
    // element i holds K = kb + group*8 + (i<8 ? i : i+8) -> two 16B runs
    // (second run at +16 halves -> offset:32 immediate).
    const int mblock = blockIdx.y * 64;
    const int aoff0 = (mblock + 0  + ncol16) * K + group * 8;
    const int aoff1 = (mblock + 16 + ncol16) * K + group * 8;
    const int aoff2 = (mblock + 32 + ncol16) * K + group * 8;
    const int aoff3 = (mblock + 48 + ncol16) * K + group * 8;

    // B fragment LDS pointers (element i reads slot group*8 + (i<8 ? i : i+8); slot p gets k_local p - krow0, odd p unfilled)
    const _Float16* bl0 = &Blds[0][(wid * 16 + ncol16) * LDS_KSTRIDE + group * 8];
    const _Float16* bl1 = &Blds[1][(wid * 16 + ncol16) * LDS_KSTRIDE + group * 8];

    v8f acc0 = {}, acc1 = {}, acc2 = {}, acc3 = {};
    union Frag { uint4 q[2]; v16h h; };

    const int nchunk = K >> 5;
    for (int chunk = 0; chunk < nchunk; ++chunk) {
        const int kb = chunk << 5;
        // ---------- cooperative B-tile fill (32 x 128 halves), branchless ----------
        _Float16* ls = (chunk & 1) ? lslot1 : lslot0;
#pragma unroll
        for (int pass = 0; pass < 16; ++pass) {
            const int kloc = krow0 + pass * 2;
            const int k = kb + kloc;
            _Float16 val;
            if (BMODE == 0) {
                const int c  = k / 9;
                const int kk = k - c * 9;
                const int ti = ftb + kk;
                const bool inb = (unsigned)ti < (unsigned)Tin;
                const int idx = fbase0 + c * cs + kk * V;
                const _Float16 ld = Bh[inb ? idx : 0];   // clamped, always-execute
                val = inb ? ld : (_Float16)0.0f;
            } else if (BMODE == 1) {
                val = Bh[fbase1 + k * cs];
            } else {
                val = (_Float16)Bf[fbase2 + k * cs];
            }
            ls[kloc] = val;
        }
        __syncthreads();

        // ---------- consume: 1 B frag (2x ds_load_b128), 4 WMMAs ----------
        Frag b;
        const _Float16* bl = (chunk & 1) ? bl1 : bl0;
        b.q[0] = *(const uint4*)(bl);
        b.q[1] = *(const uint4*)(bl + 16);

        const _Float16* wb = Wh + kb;     // uniform K advance (scalar add)
        if (chunk + 1 < nchunk) __builtin_prefetch(wb + aoff0 + 32, 0, 3);

        Frag a;
        a.q[0] = *(const uint4*)(wb + aoff0);
        a.q[1] = *(const uint4*)(wb + aoff0 + 16);
        acc0 = __builtin_amdgcn_wmma_f32_16x16x32_f16(false, a.h, false, b.h, (short)0, acc0, false, false);
        a.q[0] = *(const uint4*)(wb + aoff1);
        a.q[1] = *(const uint4*)(wb + aoff1 + 16);
        acc1 = __builtin_amdgcn_wmma_f32_16x16x32_f16(false, a.h, false, b.h, (short)0, acc1, false, false);
        a.q[0] = *(const uint4*)(wb + aoff2);
        a.q[1] = *(const uint4*)(wb + aoff2 + 16);
        acc2 = __builtin_amdgcn_wmma_f32_16x16x32_f16(false, a.h, false, b.h, (short)0, acc2, false, false);
        a.q[0] = *(const uint4*)(wb + aoff3);
        a.q[1] = *(const uint4*)(wb + aoff3 + 16);
        acc3 = __builtin_amdgcn_wmma_f32_16x16x32_f16(false, a.h, false, b.h, (short)0, acc3, false, false);
    }

    // ---------- epilogue: D row = mtile + r + group*8, col = lane&15 ----------
#pragma unroll
    for (int mt = 0; mt < 4; ++mt) {
        const v8f acc = (mt == 0) ? acc0 : (mt == 1) ? acc1 : (mt == 2) ? acc2 : acc3;
#pragma unroll
        for (int r = 0; r < 8; ++r) {
            const int mm = mblock + mt * 16 + r + group * 8;
            float y = acc[r];
            const size_t oidx = (((size_t)n * M + mm) * Tout + t) * V + v;
            if (EPI == 0) {
                y = (y + bias[mm]) * bnscale(bng[mm]) + bnb[mm];
                const size_t xb = ((size_t)n * 2) * cs + (size_t)t * V + v;
                y += rw2[mm * 2 + 0] * xbn[xb] + rw2[mm * 2 + 1] * xbn[xb + cs] + rb2[mm];
                outF[oidx] = fmaxf(y, 0.0f);
            } else if (EPI == 1) {
                y = (y + bias[mm]) * bnscale(bng[mm]) + bnb[mm];
                y += (float)resH[oidx];
                outF[oidx] = fmaxf(y, 0.0f);
            } else if (EPI == 2) {
                y = (y + bias[mm]) * bnscale(bng[mm]) + bnb[mm];
                outH[oidx] = (_Float16)fmaxf(y, 0.0f);
            } else {
                outH[oidx] = (_Float16)(y + bias[mm]);
            }
        }
    }
}

// ---------------------------------------------------------------------------
// temporal EMA smoother (in place)
// ---------------------------------------------------------------------------
__global__ void k_smooth(float* __restrict__ h, int N, int C, int T, int V) {
    int idx = blockIdx.x * blockDim.x + threadIdx.x;
    int total = N * C * V;
    if (idx >= total) return;
    int v = idx % V;
    int c = (idx / V) % C;
    int n = idx / (V * C);
    float* base = h + (((size_t)n * C + c) * T) * V + v;
    float s = base[0];
    for (int t = 1; t < T; ++t) {
        float xv = base[(size_t)t * V];
        s = SM_A * s + (1.0f - SM_A) * xv;
        base[(size_t)t * V] = s;
    }
}

__global__ void k_pool(const float* __restrict__ h, float* __restrict__ pooled,
                       int N, int C, int T, int V) {
    int idx = blockIdx.x * blockDim.x + threadIdx.x;
    if (idx >= N * C) return;
    const float* base = h + (size_t)idx * T * V;
    float s = 0.0f;
    for (int i = 0; i < T * V; ++i) s += base[i];
    pooled[idx] = s / (float)(T * V);
}

__global__ void k_fc(const float* __restrict__ pooled, const float* __restrict__ w,
                     const float* __restrict__ b, float* __restrict__ out,
                     int N, int C, int O) {
    int idx = blockIdx.x * blockDim.x + threadIdx.x;
    if (idx >= N * O) return;
    int j = idx % O;
    int n = idx / O;
    float s = b[j];
    const float* pr = pooled + (size_t)n * C;
    const float* wr = w + (size_t)j * C;
    for (int c = 0; c < C; ++c) s += pr[c] * wr[c];
    out[idx] = s;
}

// ---------------------------------------------------------------------------
extern "C" void kernel_launch(void* const* d_in, const int* in_sizes, int n_in,
                              void* d_out, int out_size, void* d_ws, size_t ws_size,
                              hipStream_t stream) {
    const int N = 128, C0 = 2, T0 = 256, V = 18;
    const float* x     = (const float*)d_in[0];
    const float* A     = (const float*)d_in[1];
    const float* dbn_g = (const float*)d_in[2];
    const float* dbn_b = (const float*)d_in[3];
    enum { GW, GB, BN1G, BN1B, TW, TB, BN2G, BN2B, RW, RB };
    const float* p[3][10];
    for (int l = 0; l < 3; ++l)
        for (int j = 0; j < 10; ++j) p[l][j] = (const float*)d_in[4 + l * 10 + j];
    const float* fc_w = (const float*)d_in[34];
    const float* fc_b = (const float*)d_in[35];
    float* outp = (float*)d_out;

    size_t off = 0;
    auto alloc = [&](size_t bytes) { size_t o = off; off = (off + bytes + 255) & ~(size_t)255; return o; };
    char* ws = (char*)d_ws;
    const size_t NTV  = (size_t)N * T0 * V;            // 589824
    const size_t E64a = 64 * NTV;                       // 37,748,736
    const size_t xbn_o = alloc(C0 * NTV * 4);
    const size_t whT1o = alloc(64 * 64 * 9 * 2);
    const size_t whG2o = alloc(128 * 64 * 2);
    const size_t whT2o = alloc(128 * 128 * 9 * 2);
    const size_t whR2o = alloc(128 * 64 * 2);
    const size_t whG3o = alloc(256 * 128 * 2);
    const size_t whT3o = alloc(256 * 256 * 9 * 2);
    const size_t whR3o = alloc(256 * 128 * 2);
    const size_t u1_o  = alloc(E64a * 2);               // u1 -> g2tmp -> g3tmp
    const size_t uBigo = alloc(2 * E64a * 2);           // u2 -> u3
    const size_t res_o = alloc(E64a * 2);               // res2 -> res3
    const size_t hA_o  = alloc(E64a * 4);               // h1 -> h3
    const size_t hB_o  = alloc(E64a * 4);               // h2
    const size_t poolo = alloc((size_t)N * 256 * 4);

    float*    xbn  = (float*)(ws + xbn_o);
    _Float16* whT1 = (_Float16*)(ws + whT1o);
    _Float16* whG2 = (_Float16*)(ws + whG2o);
    _Float16* whT2 = (_Float16*)(ws + whT2o);
    _Float16* whR2 = (_Float16*)(ws + whR2o);
    _Float16* whG3 = (_Float16*)(ws + whG3o);
    _Float16* whT3 = (_Float16*)(ws + whT3o);
    _Float16* whR3 = (_Float16*)(ws + whR3o);
    _Float16* u1   = (_Float16*)(ws + u1_o);
    _Float16* uBig = (_Float16*)(ws + uBigo);
    _Float16* res  = (_Float16*)(ws + res_o);
    float*    hA   = (float*)(ws + hA_o);
    float*    hB   = (float*)(ws + hB_o);
    float*    pool = (float*)(ws + poolo);

    auto cdiv = [](size_t a, size_t b) { return (unsigned)((a + b - 1) / b); };

    k_databn<<<cdiv(C0 * NTV, 256), 256, 0, stream>>>(x, dbn_g, dbn_b, xbn, N, C0, T0, V);

    k_w2h<<<cdiv(64 * 64 * 9, 256), 256, 0, stream>>>(p[0][TW], whT1, 64 * 64 * 9);
    k_w2h<<<cdiv(128 * 64, 256), 256, 0, stream>>>(p[1][GW], whG2, 128 * 64);
    k_w2h<<<cdiv(128 * 128 * 9, 256), 256, 0, stream>>>(p[1][TW], whT2, 128 * 128 * 9);
    k_w2h<<<cdiv(128 * 64, 256), 256, 0, stream>>>(p[1][RW], whR2, 128 * 64);
    k_w2h<<<cdiv(256 * 128, 256), 256, 0, stream>>>(p[2][GW], whG3, 256 * 128);
    k_w2h<<<cdiv(256 * 256 * 9, 256), 256, 0, stream>>>(p[2][TW], whT3, 256 * 256 * 9);
    k_w2h<<<cdiv(256 * 128, 256), 256, 0, stream>>>(p[2][RW], whR3, 256 * 128);

    k_gc1<<<cdiv(NTV, 256), 256, 0, stream>>>(xbn, A, p[0][GW], p[0][GB], p[0][BN1G],
                                              p[0][BN1B], u1, N, T0, V, 64);

    // block1 tcn + bn2 + inline residual + relu -> h1
    {
        dim3 g(cdiv(NTV, 128), 64 / 64);
        wmma_gemm<0, 0><<<g, 256, 0, stream>>>(whT1, u1, nullptr, hA, nullptr, p[0][TB],
                                               p[0][BN2G], p[0][BN2B], nullptr, p[0][RW],
                                               p[0][RB], xbn, 64, 576, 64, T0, T0, V, 1);
    }
    k_smooth<<<cdiv((size_t)N * 64 * V, 256), 256, 0, stream>>>(hA, N, 64, T0, V);

    // block2
    k_gjoint<<<cdiv(E64a, 256), 256, 0, stream>>>(hA, A, u1 /*g2tmp*/, N, 64, T0, V);
    {
        dim3 g(cdiv(NTV, 128), 128 / 64);
        wmma_gemm<1, 2><<<g, 256, 0, stream>>>(whG2, u1, nullptr, nullptr, uBig, p[1][GB],
                                               p[1][BN1G], p[1][BN1B], nullptr, nullptr,
                                               nullptr, nullptr, 128, 64, 64, T0, T0, V, 1);
    }
    {
        dim3 g(cdiv((size_t)N * 128 * V, 128), 128 / 64);
        wmma_gemm<2, 3><<<g, 256, 0, stream>>>(whR2, nullptr, hA, nullptr, res, p[1][RB],
                                               nullptr, nullptr, nullptr, nullptr, nullptr,
                                               nullptr, 128, 64, 64, T0, 128, V, 2);
    }
    {
        dim3 g(cdiv((size_t)N * 128 * V, 128), 128 / 64);
        wmma_gemm<0, 1><<<g, 256, 0, stream>>>(whT2, uBig, nullptr, hB, nullptr, p[1][TB],
                                               p[1][BN2G], p[1][BN2B], res, nullptr, nullptr,
                                               nullptr, 128, 1152, 128, T0, 128, V, 2);
    }
    k_smooth<<<cdiv((size_t)N * 128 * V, 256), 256, 0, stream>>>(hB, N, 128, 128, V);

    // block3
    k_gjoint<<<cdiv(E64a, 256), 256, 0, stream>>>(hB, A, u1 /*g3tmp*/, N, 128, 128, V);
    {
        dim3 g(cdiv((size_t)N * 128 * V, 128), 256 / 64);
        wmma_gemm<1, 2><<<g, 256, 0, stream>>>(whG3, u1, nullptr, nullptr, uBig, p[2][GB],
                                               p[2][BN1G], p[2][BN1B], nullptr, nullptr,
                                               nullptr, nullptr, 256, 128, 128, 128, 128, V, 1);
    }
    {
        dim3 g(cdiv((size_t)N * 64 * V, 128), 256 / 64);
        wmma_gemm<2, 3><<<g, 256, 0, stream>>>(whR3, nullptr, hB, nullptr, res, p[2][RB],
                                               nullptr, nullptr, nullptr, nullptr, nullptr,
                                               nullptr, 256, 128, 128, 128, 64, V, 2);
    }
    {
        dim3 g(cdiv((size_t)N * 64 * V, 128), 256 / 64);
        wmma_gemm<0, 1><<<g, 256, 0, stream>>>(whT3, uBig, nullptr, hA, nullptr, p[2][TB],
                                               p[2][BN2G], p[2][BN2B], res, nullptr, nullptr,
                                               nullptr, 256, 2304, 256, 128, 64, V, 2);
    }
    k_smooth<<<cdiv((size_t)N * 256 * V, 256), 256, 0, stream>>>(hA, N, 256, 64, V);

    k_pool<<<cdiv((size_t)N * 256, 256), 256, 0, stream>>>(hA, pool, N, 256, 64, V);
    k_fc<<<cdiv((size_t)N * 10, 256), 256, 0, stream>>>(pool, fc_w, fc_b, outp, N, 256, 10);
    (void)in_sizes; (void)n_in; (void)out_size; (void)ws_size;
}
